// AscendSparseMoeBlock_30459908063427
// MI455X (gfx1250) — matrix operand order repaired and checked by reading, not compiled, not run on hardware
//
#include <hip/hip_runtime.h>
#include <hip/hip_bf16.h>
#include <stdint.h>

#define NTOK  4096
#define HID   2048
#define INTER 1408
#define NEXP  8
#define TOPK  2
#define PAIRS (NTOK * TOPK)

#define LDS_S 72        // padded bf16 k-stride for 64-wide K tiles (144B, 16B-aligned)
#define ACT_S 72        // padded bf16 n-stride for 64-wide act staging
#define MAX_MTILES 128  // worst case: all 8192 pairs on one expert / 64 rows

typedef __attribute__((ext_vector_type(16))) __bf16 v16bf;
typedef __attribute__((ext_vector_type(8)))  __bf16 v8bf;
typedef __attribute__((ext_vector_type(8)))  float  v8f;

// ---- CDNA5 async copy: global -> LDS, 16B per lane, tracked by ASYNCcnt ----
__device__ __forceinline__ void async_copy16(void* lds_dst, const void* gsrc) {
  uint32_t l = (uint32_t)(uintptr_t)lds_dst;   // LDS aperture: low 32 bits = LDS byte addr
  uint64_t g = (uint64_t)(uintptr_t)gsrc;
  asm volatile("global_load_async_to_lds_b128 %0, %1, off" :: "v"(l), "v"(g) : "memory");
}
template <int N>
__device__ __forceinline__ void wait_async() {
  asm volatile("s_wait_asynccnt %0" :: "i"(N) : "memory");
}

// 16 fp32 held in registers for software pipelining
struct R16 { float4 a, b, c, d; };

__device__ __forceinline__ R16 ldg16(const float* src) {
  R16 r;
  r.a = *(const float4*)(src + 0);
  r.b = *(const float4*)(src + 4);
  r.c = *(const float4*)(src + 8);
  r.d = *(const float4*)(src + 12);
  return r;
}

// Convert 16 fp32 -> bf16 (v_cvt_pk_bf16_f32) and store as two 16B LDS writes.
__device__ __forceinline__ void st16(__bf16* dst, const R16& r) {
  v8f f0 = {r.a.x, r.a.y, r.a.z, r.a.w, r.b.x, r.b.y, r.b.z, r.b.w};
  v8f f1 = {r.c.x, r.c.y, r.c.z, r.c.w, r.d.x, r.d.y, r.d.z, r.d.w};
  ((v8bf*)dst)[0] = __builtin_convertvector(f0, v8bf);
  ((v8bf*)dst)[1] = __builtin_convertvector(f1, v8bf);
}

// A fragment (16x32 bf16): lane L holds M = m0+(L&15); K(j) = (j>>3)*16 + (L>>4)*8 + (j&7)
__device__ __forceinline__ v16bf load_a_frag(const __bf16* l, int m0, int ks, int lane) {
  const __bf16* p = l + (size_t)(m0 + (lane & 15)) * LDS_S + (ks << 5) + ((lane >> 4) << 3);
  v8bf lo = *(const v8bf*)p;
  v8bf hi = *(const v8bf*)(p + 16);
  v16bf r;
#pragma unroll
  for (int i = 0; i < 8; ++i) { r[i] = lo[i]; r[i + 8] = hi[i]; }
  return r;
}

// B fragment (32x16 bf16): lane L holds N = n0+(L&15); K(j) = (j>>3)*16 + (L>>4)*8 + (j&7)
__device__ __forceinline__ v16bf load_b_frag(const __bf16* l, int n0, int ks, int lane) {
  const __bf16* p = l + (size_t)(n0 + (lane & 15)) * LDS_S + (ks << 5) + ((lane >> 4) << 3);
  v8bf lo = *(const v8bf*)p;
  v8bf hi = *(const v8bf*)(p + 16);
  v16bf r;
#pragma unroll
  for (int i = 0; i < 8; ++i) { r[i] = lo[i]; r[i + 8] = hi[i]; }
  return r;
}

#define WMMA_BF16(a, b, c) \
  __builtin_amdgcn_wmma_f32_16x16x32_bf16(false, (a), false, (b), (short)0, (c), false, false)

// ---------------------------------------------------------------- zero output
__global__ __launch_bounds__(256) void zero_kernel(float4* __restrict__ p, int n4) {
  int i = blockIdx.x * blockDim.x + threadIdx.x;
  int stride = gridDim.x * blockDim.x;
  float4 z = {0.f, 0.f, 0.f, 0.f};
  for (; i < n4; i += stride) p[i] = z;
}

// ---------------------------------------------------------------- routing
__global__ __launch_bounds__(256) void route_kernel(
    const int* __restrict__ ids, const float* __restrict__ tw,
    int* __restrict__ seg, int* __restrict__ row_token, float* __restrict__ row_weight) {
  __shared__ int s_cnt[NEXP];
  __shared__ int s_off[NEXP];
  __shared__ int s_cur[NEXP];
  int tid = threadIdx.x;
  if (tid < NEXP) { s_cnt[tid] = 0; s_cur[tid] = 0; }
  __syncthreads();
  for (int p = tid; p < PAIRS; p += 256) atomicAdd(&s_cnt[ids[p]], 1);
  __syncthreads();
  if (tid == 0) {
    int run = 0;
    for (int e = 0; e < NEXP; ++e) { s_off[e] = run; seg[e] = run; run += s_cnt[e]; }
    seg[NEXP] = run;
  }
  __syncthreads();
  for (int p = tid; p < PAIRS; p += 256) {
    int e = ids[p];
    int pos = s_off[e] + atomicAdd(&s_cur[e], 1);
    row_token[pos]  = p >> 1;
    row_weight[pos] = tw[p];
  }
}

// ------------------------------------------- gate_up GEMM + SwiGLU -> act(bf16)
__global__ __launch_bounds__(256, 2) void moe_gateup_kernel(
    const float* __restrict__ x, const float* __restrict__ w1,
    const int* __restrict__ seg, const int* __restrict__ row_token,
    __bf16* __restrict__ act) {
  int e  = blockIdx.y >> 7;
  int mt = blockIdx.y & (MAX_MTILES - 1);
  int s0 = seg[e], s1 = seg[e + 1];
  int row0 = s0 + mt * 64;
  if (row0 >= s1) return;
  int n0 = blockIdx.x * 64;

  __shared__ __align__(16) __bf16 la  [64 * LDS_S];
  __shared__ __align__(16) __bf16 lbg [64 * LDS_S];
  __shared__ __align__(16) __bf16 lbu [64 * LDS_S];
  __shared__ __align__(16) __bf16 lact[64 * ACT_S];

  int tid = threadIdx.x;
  int lane = tid & 31, wave = tid >> 5;       // wave32
  int msub = wave & 3;                        // 4 m-subtiles of 16
  int nb   = (wave >> 2) << 1;                // waves 0-3 -> n 0,1 ; 4-7 -> n 2,3

  int srow = tid >> 2;                        // 0..63
  int scol = (tid & 3) << 4;                  // 0,16,32,48 (floats)
  int arow = row0 + srow; if (arow >= s1) arow = s1 - 1;   // clamp (masked later)
  const float* aptr = x  + (size_t)row_token[arow] * HID + scol;
  const float* w1e  = w1 + (size_t)e * (2 * INTER) * HID;
  const float* gptr = w1e + (size_t)(n0 + srow) * HID + scol;
  const float* uptr = w1e + (size_t)(INTER + n0 + srow) * HID + scol;

  __bf16* sa = la  + srow * LDS_S + scol;
  __bf16* sg = lbg + srow * LDS_S + scol;
  __bf16* su = lbu + srow * LDS_S + scol;

  v8f cg0 = {}, cg1 = {}, cu0 = {}, cu1 = {};

  // software pipeline: stage kt lives in registers while stage kt-64 computes
  R16 ra = ldg16(aptr), rg = ldg16(gptr), ru = ldg16(uptr);

  for (int kt = 0; kt < HID; kt += 64) {      // 32 stages, 2 WMMA K-steps each
    st16(sa, ra);
    st16(sg, rg);
    st16(su, ru);
    __syncthreads();
    if (kt + 64 < HID) {                      // prefetch next stage under the WMMAs
      ra = ldg16(aptr + kt + 64);
      rg = ldg16(gptr + kt + 64);
      ru = ldg16(uptr + kt + 64);
    }
#pragma unroll
    for (int ks = 0; ks < 2; ++ks) {
      v16bf a   = load_a_frag(la,  msub << 4, ks, lane);
      v16bf bg0 = load_b_frag(lbg, (nb + 0) << 4, ks, lane);
      v16bf bg1 = load_b_frag(lbg, (nb + 1) << 4, ks, lane);
      v16bf bu0 = load_b_frag(lbu, (nb + 0) << 4, ks, lane);
      v16bf bu1 = load_b_frag(lbu, (nb + 1) << 4, ks, lane);
      cg0 = WMMA_BF16(a, bg0, cg0);
      cg1 = WMMA_BF16(a, bg1, cg1);
      cu0 = WMMA_BF16(a, bu0, cu0);
      cu1 = WMMA_BF16(a, bu1, cu1);
    }
    __syncthreads();
  }

  // SwiGLU: silu(g)*u = g*u*rcp(1+exp(-g)); stage [m][n] in LDS for coalesced stores
  {
    int n_lane = lane & 15;
    int mrow   = (msub << 4) + ((lane >> 4) << 3);
#pragma unroll
    for (int j = 0; j < 8; ++j) {
      float g0 = cg0[j], u0 = cu0[j];
      float g1 = cg1[j], u1 = cu1[j];
      float a0 = g0 * u0 * __builtin_amdgcn_rcpf(1.f + __expf(-g0));
      float a1 = g1 * u1 * __builtin_amdgcn_rcpf(1.f + __expf(-g1));
      lact[(mrow + j) * ACT_S + ((nb + 0) << 4) + n_lane] = (__bf16)a0;
      lact[(mrow + j) * ACT_S + ((nb + 1) << 4) + n_lane] = (__bf16)a1;
    }
  }
  __syncthreads();
  {
    int m = tid >> 2;
    int c = (tid & 3) << 4;                   // 16 bf16 = 32B per thread
    int grow = row0 + m;
    if (grow < s1) {
      const uint4* s = (const uint4*)(lact + m * ACT_S + c);
      uint4* d = (uint4*)(act + (size_t)grow * INTER + n0 + c);
      d[0] = s[0]; d[1] = s[1];
    }
  }
}

// -------------------------------- down GEMM + weighted atomic scatter-combine
__global__ __launch_bounds__(256, 2) void moe_down_kernel(
    const __bf16* __restrict__ act, const float* __restrict__ w2,
    const int* __restrict__ seg, const int* __restrict__ row_token,
    const float* __restrict__ row_weight, float* __restrict__ out) {
  int e  = blockIdx.y >> 7;
  int mt = blockIdx.y & (MAX_MTILES - 1);
  int s0 = seg[e], s1 = seg[e + 1];
  int row0 = s0 + mt * 64;
  if (row0 >= s1) return;
  int h0 = blockIdx.x * 64;

  __shared__ __align__(16) __bf16 la0[64 * LDS_S];   // ping-pong A (async-filled)
  __shared__ __align__(16) __bf16 la1[64 * LDS_S];
  __shared__ __align__(16) __bf16 lb [64 * LDS_S];

  int tid = threadIdx.x, lane = tid & 31, wave = tid >> 5;
  int msub = wave & 3, nb = (wave >> 2) << 1;

  int srow = tid >> 2, scol = (tid & 3) << 4;
  int arow = row0 + srow; if (arow >= s1) arow = s1 - 1;
  const __bf16* aptr = act + (size_t)arow * INTER + scol;        // already bf16
  const float*  bptr = w2 + ((size_t)e * HID + h0 + srow) * INTER + scol;

  int soff = srow * LDS_S + scol;
  __bf16* sb = lb + soff;

  v8f c0 = {}, c1 = {};

  // prologue: async-fill A buffer 0, prefetch B regs for stage 0
  async_copy16(la0 + soff,     aptr);
  async_copy16(la0 + soff + 8, aptr + 8);
  R16 rb = ldg16(bptr);

  for (int kt = 0; kt < INTER; kt += 64) {                       // 22 stages
    int cur = (kt >> 6) & 1;
    __bf16* laCur = cur ? la1 : la0;        // ternary select: no LDS-pointer arrays
    __bf16* laNxt = cur ? la0 : la1;
    st16(sb, rb);
    if (kt + 64 < INTER) {
      // async-fill the inactive A buffer for the next stage; in-order completion
      // means waiting ASYNCcnt<=2 retires the pair that filled the current buffer
      async_copy16(laNxt + soff,     aptr + kt + 64);
      async_copy16(laNxt + soff + 8, aptr + kt + 72);
      wait_async<2>();
    } else {
      wait_async<0>();
    }
    __syncthreads();
    if (kt + 64 < INTER) rb = ldg16(bptr + kt + 64);             // hide under WMMAs
#pragma unroll
    for (int ks = 0; ks < 2; ++ks) {
      v16bf a  = load_a_frag(laCur, msub << 4, ks, lane);
      v16bf b0 = load_b_frag(lb, (nb + 0) << 4, ks, lane);
      v16bf b1 = load_b_frag(lb, (nb + 1) << 4, ks, lane);
      c0 = WMMA_BF16(a, b0, c0);
      c1 = WMMA_BF16(a, b1, c1);
    }
    __syncthreads();
  }

  int ncol0 = h0 + ((nb + 0) << 4) + (lane & 15);
  int ncol1 = h0 + ((nb + 1) << 4) + (lane & 15);
  int mrow  = (msub << 4) + ((lane >> 4) << 3);
#pragma unroll
  for (int j = 0; j < 8; ++j) {
    int row = row0 + mrow + j;
    if (row < s1) {
      int   t = row_token[row];
      float w = row_weight[row];
      float* obase = out + (size_t)t * HID;
      unsafeAtomicAdd(obase + ncol0, w * c0[j]);   // global_atomic_add_f32
      unsafeAtomicAdd(obase + ncol1, w * c1[j]);
    }
  }
}

// ---------------------------------------------------------------------- launch
extern "C" void kernel_launch(void* const* d_in, const int* in_sizes, int n_in,
                              void* d_out, int out_size, void* d_ws, size_t ws_size,
                              hipStream_t stream) {
  const float* x   = (const float*)d_in[0];   // (4096, 2048)
  const float* w1  = (const float*)d_in[1];   // (8, 2816, 2048)
  const float* w2  = (const float*)d_in[2];   // (8, 2048, 1408)
  const float* tw  = (const float*)d_in[3];   // (4096, 2)
  const int*   ids = (const int*)d_in[4];     // (4096, 2)
  float* out = (float*)d_out;

  uint8_t* ws = (uint8_t*)d_ws;
  int*    seg        = (int*)(ws + 0);                          // 9 ints
  int*    row_token  = (int*)(ws + 256);                        // 8192 ints
  float*  row_weight = (float*)(ws + 256 + PAIRS * 4);          // 8192 floats
  __bf16* act        = (__bf16*)(ws + 256 + PAIRS * 8 + 256);   // 8192x1408 bf16 (16B-aligned)

  zero_kernel<<<4096, 256, 0, stream>>>((float4*)out, (NTOK * HID) / 4);
  route_kernel<<<1, 256, 0, stream>>>(ids, tw, seg, row_token, row_weight);
  moe_gateup_kernel<<<dim3(INTER / 64, NEXP * MAX_MTILES), 256, 0, stream>>>(
      x, w1, seg, row_token, act);
  moe_down_kernel<<<dim3(HID / 64, NEXP * MAX_MTILES), 256, 0, stream>>>(
      act, w2, seg, row_token, row_weight, out);
}
